// Attention_42717744726584
// MI455X (gfx1250) — compile-verified, matrix operand order checked by reading, not run on hardware
//
#include <hip/hip_runtime.h>
#include <hip/hip_bf16.h>
#include <stdint.h>

// ---------------------------------------------------------------------------
// MI455X (gfx1250) attention layer, f16 WMMA + async-LDS + flash attention.
// S=2048, D=4096, H=32, hd=128.  Compute-bound (~344 GFLOP vs ~14us of HBM).
// GEMM: 128x256 block tile, 8 waves, 64x64 wave tile -> 16 WMMA per 8 LDS
// fragment loads per 32-deep k-step; double-buffered async global->LDS fill.
// ---------------------------------------------------------------------------

constexpr int S   = 2048;
constexpr int DIM = 4096;
constexpr int H   = 32;
constexpr int HD  = 128;

typedef __attribute__((ext_vector_type(16))) _Float16 v16h;
typedef __attribute__((ext_vector_type(8)))  _Float16 v8h;
typedef __attribute__((ext_vector_type(8)))  float    v8f;

// ---- CDNA5 async global->LDS (ASYNCcnt-tracked) ---------------------------
__device__ __forceinline__ void async_b128(uint32_t lds_addr, const void* g) {
    asm volatile("global_load_async_to_lds_b128 %0, %1, off"
                 :: "v"(lds_addr), "v"((unsigned long long)(uintptr_t)g)
                 : "memory");
}
__device__ __forceinline__ void wait_async0() {
    asm volatile("s_wait_asynccnt 0" ::: "memory");
}
__device__ __forceinline__ void wait_async6() {
    asm volatile("s_wait_asynccnt 6" ::: "memory");
}

// ---- WMMA fragment helpers (16-bit A/B operand layout, wave32) ------------
// lane = (r, hi): r = lane&15 selects row, hi = lane>>4 selects K-halves
// {0..7,16..23} vs {8..15,24..31}.  Two contiguous 16B loads per fragment.
__device__ __forceinline__ v16h combine16(v8h lo, v8h hi) {
    return __builtin_shufflevector(lo, hi, 0,1,2,3,4,5,6,7,8,9,10,11,12,13,14,15);
}
__device__ __forceinline__ v16h ldfrag(const _Float16* p) {
    v8h lo = *(const v8h*)p;
    v8h hi = *(const v8h*)(p + 16);
    return combine16(lo, hi);
}
__device__ __forceinline__ v8f wmma_f16(v16h a, v16h b, v8f c) {
    return __builtin_amdgcn_wmma_f32_16x16x32_f16(false, a, false, b,
                                                  (short)0, c, false, false);
}

// ===========================================================================
// Kernel 1: f32 -> f16 elementwise convert (x)
// ===========================================================================
__global__ void cvt_f32_f16(const float* __restrict__ src,
                            _Float16* __restrict__ dst, int n) {
    int i = blockIdx.x * blockDim.x + threadIdx.x;
    if (i < n) dst[i] = (_Float16)src[i];
}

// ===========================================================================
// Kernel 2: f32 -> f16 convert + transpose (weights: W[k][n] -> Wt[n][k])
// so GEMM B-operand fragments are contiguous along K.
// ===========================================================================
__global__ __launch_bounds__(256)
void cvt_transpose(const float* __restrict__ src, _Float16* __restrict__ dst,
                   int K, int N) {
    __shared__ float tile[32][33];
    int bx = blockIdx.x * 32;  // n base
    int by = blockIdx.y * 32;  // k base
    int tx = threadIdx.x, ty = threadIdx.y;        // 32 x 8
    #pragma unroll
    for (int i = 0; i < 32; i += 8)
        tile[ty + i][tx] = src[(size_t)(by + ty + i) * N + bx + tx];
    __syncthreads();
    #pragma unroll
    for (int i = 0; i < 32; i += 8)
        dst[(size_t)(bx + ty + i) * K + by + tx] = (_Float16)tile[tx][ty + i];
}

// ===========================================================================
// Kernel 3: GEMM  C[M,N] = A[M,K] (f16, row) x Bt[N,K] (f16, row = B^T)
// Block tile 128x256, BK=32; 8 waves (2x4), wave tile 64x64 = 16 WMMA accums.
// Double-buffered LDS filled by async b128 loads (6 per thread per tile).
// OMODE: 0 = f16 row-major, 1 = f16 transposed (C^T[N,M]), 2 = f32 row-major
// ===========================================================================
template <int OMODE>
__global__ __launch_bounds__(256)
void gemm_f16(const _Float16* __restrict__ A, const _Float16* __restrict__ Bt,
              void* __restrict__ Cv, int M, int N, int K) {
    constexpr int BM = 128, BN = 256, BK = 32, BKP = 40;  // pad vs bank conflict
    __shared__ _Float16 sA[2][BM * BKP];
    __shared__ _Float16 sB[2][BN * BKP];

    const int tid  = threadIdx.x;
    const int lane = tid & 31;
    const int wave = tid >> 5;
    const int r  = lane & 15;
    const int hi = lane >> 4;
    const int wm = wave >> 2;      // 0..1  -> 64-row slab
    const int wn = wave & 3;       // 0..3  -> 64-col slab
    const size_t mbase = (size_t)blockIdx.y * BM;
    const size_t nbase = (size_t)blockIdx.x * BN;

    v8f acc[4][4];
    #pragma unroll
    for (int i = 0; i < 4; ++i)
        #pragma unroll
        for (int j = 0; j < 4; ++j) acc[i][j] = (v8f)0.0f;

    auto issue_tile = [&](int buf, int kb) {
        const size_t k0 = (size_t)kb * BK;
        // A tile: 128 rows x 64B = 512 16B-chunks; 2 per thread
        #pragma unroll
        for (int c = tid; c < 512; c += 256) {
            int row = c >> 2, q = c & 3;
            async_b128((uint32_t)(uintptr_t)&sA[buf][row * BKP + q * 8],
                       A + (mbase + row) * K + k0 + q * 8);
        }
        // B tile: 256 rows x 64B = 1024 16B-chunks; 4 per thread
        #pragma unroll
        for (int c = tid; c < 1024; c += 256) {
            int row = c >> 2, q = c & 3;
            async_b128((uint32_t)(uintptr_t)&sB[buf][row * BKP + q * 8],
                       Bt + (nbase + row) * K + k0 + q * 8);
        }
    };

    auto compute = [&](int buf) {
        v16h af[4], bf[4];
        #pragma unroll
        for (int i = 0; i < 4; ++i)
            af[i] = ldfrag(&sA[buf][(wm * 64 + i * 16 + r) * BKP + hi * 8]);
        #pragma unroll
        for (int j = 0; j < 4; ++j)
            bf[j] = ldfrag(&sB[buf][(wn * 64 + j * 16 + r) * BKP + hi * 8]);
        #pragma unroll
        for (int i = 0; i < 4; ++i)
            #pragma unroll
            for (int j = 0; j < 4; ++j)
                acc[i][j] = wmma_f16(af[i], bf[j], acc[i][j]);
    };

    const int NK = K / BK;
    issue_tile(0, 0);
    for (int kb = 0; kb < NK; ++kb) {
        if (kb + 1 < NK) { issue_tile((kb + 1) & 1, kb + 1); wait_async6(); }
        else             { wait_async0(); }
        __syncthreads();
        compute(kb & 1);
        __syncthreads();
    }

    // Epilogue.  C-tile layout: lane holds col n = r, rows m = hi*8 + e.
    #pragma unroll
    for (int i = 0; i < 4; ++i) {
        #pragma unroll
        for (int j = 0; j < 4; ++j) {
            const size_t col = nbase + wn * 64 + j * 16 + r;
            const size_t m0  = mbase + wm * 64 + i * 16 + hi * 8;
            if constexpr (OMODE == 0) {
                _Float16* C = (_Float16*)Cv;
                #pragma unroll
                for (int e = 0; e < 8; ++e)
                    C[(m0 + e) * N + col] = (_Float16)acc[i][j][e];
            } else if constexpr (OMODE == 1) {
                _Float16* Ct = (_Float16*)Cv;  // [N][M]
                v8h hv;
                #pragma unroll
                for (int e = 0; e < 8; ++e) hv[e] = (_Float16)acc[i][j][e];
                *(v8h*)(Ct + col * M + m0) = hv;   // 16B contiguous store
            } else {
                float* C = (float*)Cv;
                #pragma unroll
                for (int e = 0; e < 8; ++e)
                    C[(m0 + e) * N + col] = acc[i][j][e];
            }
        }
    }
}

// ===========================================================================
// Kernel 4: RoPE on Q/K (interleaved pairs within each head dim)
// ===========================================================================
__global__ void rope_kernel(_Float16* __restrict__ t,
                            const float* __restrict__ cosb,
                            const float* __restrict__ sinb, int npairs) {
    int idx = blockIdx.x * blockDim.x + threadIdx.x;
    if (idx >= npairs) return;
    int s   = idx / (DIM / 2);
    int rem = idx - s * (DIM / 2);
    int i   = rem & (HD / 2 - 1);           // pair index within head
    float c  = cosb[s * (HD / 2) + i];
    float sn = sinb[s * (HD / 2) + i];
    _Float16* p = t + (size_t)idx * 2;
    float re = (float)p[0], im = (float)p[1];
    p[0] = (_Float16)(re * c - im * sn);
    p[1] = (_Float16)(re * sn + im * c);
}

// ===========================================================================
// Kernel 5: flash attention.  Block = 4 waves x 16 q-rows = 64 rows, per head.
// Each wave streams causal kv tiles of 32: 8 score-WMMAs worth of K, online
// softmax via lane shuffles, P restaged through LDS as A-operand, 8 PV WMMAs
// into a 16x128 f32 accumulator.  V comes in pre-transposed (Vt[D][S]).
// ===========================================================================
__global__ __launch_bounds__(128)
void flash_attn(const _Float16* __restrict__ qh, const _Float16* __restrict__ kh,
                const _Float16* __restrict__ vt, _Float16* __restrict__ att) {
    const int h     = blockIdx.y;
    const int wave  = threadIdx.x >> 5;
    const int lane  = threadIdx.x & 31;
    const int r     = lane & 15;
    const int hi    = lane >> 4;
    const int qbase = blockIdx.x * 64 + wave * 16;

    __shared__ _Float16 pls[4][16 * 40];
    _Float16* myp = &pls[wave][0];

    // Q fragments for d = 0,32,64,96 (resident for whole kv sweep)
    v16h qf[4];
    {
        const _Float16* qrow = qh + (size_t)(qbase + r) * DIM + h * HD + hi * 8;
        #pragma unroll
        for (int d4 = 0; d4 < 4; ++d4) qf[d4] = ldfrag(qrow + d4 * 32);
    }

    v8f oacc[8];
    #pragma unroll
    for (int j = 0; j < 8; ++j) oacc[j] = (v8f)0.0f;
    float m8[8], l8[8];
    #pragma unroll
    for (int i = 0; i < 8; ++i) { m8[i] = -3.0e38f; l8[i] = 0.0f; }

    const float scale = 0.08838834764831845f;  // 1/sqrt(128)

    for (int kv = 0; kv < qbase + 16; kv += 32) {
        // ---- scores: 16 x 32 tile = two 16x16 accumulators -----------------
        v8f s0 = (v8f)0.0f, s1 = (v8f)0.0f;
        {
            const _Float16* k0 = kh + (size_t)(kv + r) * DIM + h * HD + hi * 8;
            const _Float16* k1 = k0 + (size_t)16 * DIM;
            #pragma unroll
            for (int d4 = 0; d4 < 4; ++d4) {
                s0 = wmma_f16(qf[d4], ldfrag(k0 + d4 * 32), s0);
                s1 = wmma_f16(qf[d4], ldfrag(k1 + d4 * 32), s1);
            }
        }
        const bool diag = (kv + 31 > qbase);

        // ---- online softmax (row stats live redundantly across 16 lanes) ---
        #pragma unroll
        for (int i = 0; i < 8; ++i) {
            const int row = qbase + hi * 8 + i;
            float a0 = s0[i] * scale;
            float a1 = s1[i] * scale;
            if (diag) {
                if (kv + r > row)      a0 = -3.0e38f;
                if (kv + 16 + r > row) a1 = -3.0e38f;
            }
            float mx = fmaxf(a0, a1);
            #pragma unroll
            for (int m = 1; m < 16; m <<= 1) mx = fmaxf(mx, __shfl_xor(mx, m, 32));
            const float mnew = fmaxf(m8[i], mx);
            const float corr = __expf(m8[i] - mnew);
            const float p0 = __expf(a0 - mnew);
            const float p1 = __expf(a1 - mnew);
            float rs = p0 + p1;
            #pragma unroll
            for (int m = 1; m < 16; m <<= 1) rs += __shfl_xor(rs, m, 32);
            l8[i] = l8[i] * corr + rs;
            m8[i] = mnew;
            #pragma unroll
            for (int j = 0; j < 8; ++j) oacc[j][i] *= corr;
            // stage P into LDS (C-tile layout -> A-operand layout)
            myp[(hi * 8 + i) * 40 + r]      = (_Float16)p0;
            myp[(hi * 8 + i) * 40 + r + 16] = (_Float16)p1;
        }

        // ---- PV: P(16x32) x V(32x128) via Vt fragments ---------------------
        v16h pf = ldfrag(myp + r * 40 + hi * 8);
        #pragma unroll
        for (int j = 0; j < 8; ++j) {
            const _Float16* vp =
                vt + (size_t)(h * HD + j * 16 + r) * S + kv + hi * 8;
            oacc[j] = wmma_f16(pf, ldfrag(vp), oacc[j]);
        }
    }

    // ---- epilogue: O /= l, write f16 ---------------------------------------
    #pragma unroll
    for (int j = 0; j < 8; ++j) {
        const size_t col = (size_t)h * HD + j * 16 + r;
        #pragma unroll
        for (int i = 0; i < 8; ++i) {
            const size_t row = qbase + hi * 8 + i;
            att[row * DIM + col] = (_Float16)(oacc[j][i] / l8[i]);
        }
    }
}

// ===========================================================================
// Host-side orchestration
// ===========================================================================
extern "C" void kernel_launch(void* const* d_in, const int* in_sizes, int n_in,
                              void* d_out, int out_size, void* d_ws, size_t ws_size,
                              hipStream_t stream) {
    const float* x    = (const float*)d_in[0];
    const float* wq   = (const float*)d_in[1];
    const float* wk   = (const float*)d_in[2];
    const float* wv   = (const float*)d_in[3];
    const float* wo   = (const float*)d_in[4];
    const float* fcos = (const float*)d_in[5];
    const float* fsin = (const float*)d_in[6];
    // d_in[7] (mask) and d_in[8] (start_pos) handled implicitly (causal, 0).
    float* out = (float*)d_out;

    const size_t SD = (size_t)S * DIM;    // 8Mi elements
    const size_t DD = (size_t)DIM * DIM;  // 16Mi elements

    char* w = (char*)d_ws;
    size_t off = 0;
    _Float16* xh  = (_Float16*)(w + off); off += SD * 2;
    _Float16* wqt = (_Float16*)(w + off); off += DD * 2;
    _Float16* wkt = (_Float16*)(w + off); off += DD * 2;
    _Float16* wvt = (_Float16*)(w + off); off += DD * 2;
    _Float16* wot = (_Float16*)(w + off); off += DD * 2;
    _Float16* qhb = (_Float16*)(w + off); off += SD * 2;
    _Float16* khb = (_Float16*)(w + off); off += SD * 2;
    _Float16* vtb = (_Float16*)(w + off); off += SD * 2;   // [DIM][S]
    _Float16* att = (_Float16*)(w + off); off += SD * 2;

    // 1) precision downcast
    cvt_f32_f16<<<(int)(SD / 256), 256, 0, stream>>>(x, xh, (int)SD);
    dim3 tg(DIM / 32, DIM / 32), tb(32, 8);
    cvt_transpose<<<tg, tb, 0, stream>>>(wq, wqt, DIM, DIM);
    cvt_transpose<<<tg, tb, 0, stream>>>(wk, wkt, DIM, DIM);
    cvt_transpose<<<tg, tb, 0, stream>>>(wv, wvt, DIM, DIM);
    cvt_transpose<<<tg, tb, 0, stream>>>(wo, wot, DIM, DIM);

    // 2) QKV projections (V written transposed for PV B-operand)
    dim3 gg(DIM / 256, S / 128);
    gemm_f16<0><<<gg, 256, 0, stream>>>(xh, wqt, qhb, S, DIM, DIM);
    gemm_f16<0><<<gg, 256, 0, stream>>>(xh, wkt, khb, S, DIM, DIM);
    gemm_f16<1><<<gg, 256, 0, stream>>>(xh, wvt, vtb, S, DIM, DIM);

    // 3) RoPE on Q and K
    const int npairs = (int)(SD / 2);
    rope_kernel<<<(npairs + 255) / 256, 256, 0, stream>>>(qhb, fcos, fsin, npairs);
    rope_kernel<<<(npairs + 255) / 256, 256, 0, stream>>>(khb, fcos, fsin, npairs);

    // 4) flash attention
    flash_attn<<<dim3(S / 64, H), 128, 0, stream>>>(qhb, khb, vtb, att);

    // 5) output projection (f32 result)
    gemm_f16<2><<<gg, 256, 0, stream>>>(att, wot, out, S, DIM, DIM);
}
